// DGCNN_37288906064253
// MI455X (gfx1250) — matrix operand order repaired and checked by reading, not compiled, not run on hardware
//
#include <hip/hip_runtime.h>
#include <hip/hip_bf16.h>

typedef __attribute__((ext_vector_type(16))) _Float16 v16h;
typedef __attribute__((ext_vector_type(8)))  _Float16 v8h;
typedef __attribute__((ext_vector_type(8)))  float    v8f;

#define B_  32
#define N_  2048
#define KNN 20

__device__ __forceinline__ float lrelu(float x){ return x >= 0.f ? x : 0.2f * x; }

__device__ __forceinline__ v16h cat2(v8h lo, v8h hi){
  v16h r;
#pragma unroll
  for (int i = 0; i < 8; ++i){ r[i] = lo[i]; r[i+8] = hi[i]; }
  return r;
}

__device__ __forceinline__ v8f wmma_f16(v16h a, v16h b, v8f c){
  // D = A(16x32 f16) * B(32x16 f16) + C(16x16 f32)
  return __builtin_amdgcn_wmma_f32_16x16x32_f16(false, a, false, b, (short)0, c, false, false);
}

// ---------------------------------------------------------------------------
// Kernel 1: 3-D kNN (top-20 by neg squared distance). One thread per point,
// batch positions staged in LDS.
// ---------------------------------------------------------------------------
__global__ __launch_bounds__(256) void k_knn3(const float* __restrict__ x, int* __restrict__ idx){
  __shared__ float sp[N_ * 3];
  __shared__ float sq[N_];
  int b = blockIdx.x >> 3;
  int chunk = blockIdx.x & 7;
  const float* xb = x + (size_t)b * N_ * 6;
  for (int i = threadIdx.x; i < N_; i += 256){
    float px = xb[i*6+0], py = xb[i*6+1], pz = xb[i*6+2];
    sp[i*3+0] = px; sp[i*3+1] = py; sp[i*3+2] = pz;
    sq[i] = px*px + py*py + pz*pz;
  }
  __syncthreads();
  int n = chunk * 256 + threadIdx.x;
  float cx = sp[n*3], cy = sp[n*3+1], cz = sp[n*3+2], cs = sq[n];
  float bv[KNN]; int bi[KNN];
#pragma unroll
  for (int k = 0; k < KNN; ++k){ bv[k] = -3.4e38f; bi[k] = 0; }
  for (int m = 0; m < N_; ++m){
    float d = 2.f*(cx*sp[m*3] + cy*sp[m*3+1] + cz*sp[m*3+2]) - cs - sq[m];
    if (d > bv[KNN-1]){
      int p = KNN - 1;
      while (p > 0 && bv[p-1] < d){ bv[p] = bv[p-1]; bi[p] = bi[p-1]; --p; }
      bv[p] = d; bi[p] = m;
    }
  }
  int* out = idx + ((size_t)b * N_ + n) * KNN;
  for (int k = 0; k < KNN; ++k) out[k] = bi[k];
}

// ---------------------------------------------------------------------------
// Kernel 2: edge-conv 1 (6 -> 64) + max over K via P+Q factorization folded
// per-thread. Writes x1 (f16, first half of x12h) and sq-norm of x1.
// ---------------------------------------------------------------------------
__global__ __launch_bounds__(256) void k_edge1(const float* __restrict__ x, const float* __restrict__ W1,
                                               const int* __restrict__ idx,
                                               _Float16* __restrict__ x12h, float* __restrict__ sq1){
  __shared__ float w[64 * 6];
  __shared__ float red[256];
  for (int i = threadIdx.x; i < 64*6; i += 256) w[i] = W1[i];
  __syncthreads();
  int o  = threadIdx.x & 63;
  int pt = threadIdx.x >> 6;
  int b  = blockIdx.x >> 9;
  int n  = (blockIdx.x & 511) * 4 + pt;
  const float* xb = x + (size_t)b * N_ * 6;
  float cx = xb[n*6], cy = xb[n*6+1], cz = xb[n*6+2];
  const int* id = idx + ((size_t)b * N_ + n) * KNN;
  float w0 = w[o*6+0], w1 = w[o*6+1], w2 = w[o*6+2];
  float w3 = w[o*6+3], w4 = w[o*6+4], w5 = w[o*6+5];
  float base = (w3 - w0)*cx + (w4 - w1)*cy + (w5 - w2)*cz;
  float best = -3.4e38f;
  for (int k = 0; k < KNN; ++k){
    int m = id[k];
    float v = w0*xb[m*6] + w1*xb[m*6+1] + w2*xb[m*6+2] + base;
    best = fmaxf(best, v);
  }
  float val = lrelu(best);                 // max_k lrelu == lrelu max_k (monotone)
  x12h[((size_t)b * N_ + n) * 128 + o] = (_Float16)val;
  red[threadIdx.x] = val * val;
  __syncthreads();
  for (int s = 32; s > 0; s >>= 1){
    if (o < s) red[threadIdx.x] += red[threadIdx.x + s];
    __syncthreads();
  }
  if (o == 0) sq1[(size_t)b * N_ + n] = red[threadIdx.x];
}

// ---------------------------------------------------------------------------
// Kernel 3: 64-dim kNN. One wave per 16-row panel. Gram tile via WMMA f16
// (K=64 -> 2 WMMA), register double-buffered B tiles (loads issued one tile
// ahead so the scan hides their latency), sq[] cached in LDS and folded into
// the tile at store time. All 32 lanes scan: lane l and l+16 split each
// row's 16 columns, private top-20s merged at the end (2-way sorted merge).
// ---------------------------------------------------------------------------
__global__ __launch_bounds__(32) void k_knn64(const _Float16* __restrict__ x12h,
                                              const float* __restrict__ sq1,
                                              int* __restrict__ idx){
  __shared__ float sqs[N_];          // 8KB: batch sq-norms
  __shared__ float tile[16 * 16];    // 2*G - sq[col]
  __shared__ float mv[32 * KNN];
  __shared__ int   mi[32 * KNN];
  int b = blockIdx.x >> 7;
  int mbase = (blockIdx.x & 127) << 4;
  int l = threadIdx.x;
  const _Float16* xb = x12h + (size_t)b * N_ * 128;
  const float* sqb = sq1 + (size_t)b * N_;

  for (int i = l * 4; i < N_; i += 32 * 4)
    *(float4*)&sqs[i] = *(const float4*)&sqb[i];

  // A tile (rows fixed). Contiguous K order matching B: lane<16 K{0..15},
  // lane>=16 K{16..31} of row (lane&15).
  int mrow = mbase + (l & 15);
  const _Float16* arow = xb + (size_t)mrow * 128;
  int asel = (l >> 4) << 4;
  v16h a0 = cat2(*(const v8h*)(arow +  0 + asel), *(const v8h*)(arow +  8 + asel));
  v16h a1 = cat2(*(const v8h*)(arow + 32 + asel), *(const v8h*)(arow + 40 + asel));
  __syncthreads();

  float bv[KNN]; int bi[KNN];
#pragma unroll
  for (int k = 0; k < KNN; ++k){ bv[k] = -3.4e38f; bi[k] = 0; }
  float rowsq = sqs[mbase + (l & 15)];
  int r0   = (l >> 4) << 3;     // C rows this lane holds
  int bsel = (l >> 4) << 4;     // B K-split
  int ch   = (l >> 4) << 3;     // scan column half

  auto loadB = [&](int t, v16h& o0, v16h& o1){
    const _Float16* brow = xb + (size_t)((t << 4) + (l & 15)) * 128;
    o0 = *(const v16h*)(brow +  0 + bsel);
    o1 = *(const v16h*)(brow + 32 + bsel);
  };
  v16h cb0, cb1;
  loadB(0, cb0, cb1);

  for (int t = 0; t < 128; ++t){
    v16h nb0, nb1;
    loadB((t + 1) & 127, nb0, nb1);           // issue next tile's loads now
    v8f c = {};
    c = wmma_f16(a0, cb0, c);
    c = wmma_f16(a1, cb1, c);
    float sqc = sqs[(t << 4) + (l & 15)];     // this lane's C column
#pragma unroll
    for (int r = 0; r < 8; ++r) tile[(r0 + r) * 16 + (l & 15)] = 2.f * c[r] - sqc;
    __syncthreads();
    {
      int rbase = (l & 15) * 16 + ch;
      int mb = (t << 4) + ch;
      for (int j = 0; j < 8; ++j){
        float d = tile[rbase + j] - rowsq;
        if (d > bv[KNN-1]){
          int p = KNN - 1;
          while (p > 0 && bv[p-1] < d){ bv[p] = bv[p-1]; bi[p] = bi[p-1]; --p; }
          bv[p] = d; bi[p] = mb + j;
        }
      }
    }
    __syncthreads();
    cb0 = nb0; cb1 = nb1;
  }
  // merge the two half-scans of each row
#pragma unroll
  for (int k = 0; k < KNN; ++k){ mv[l*KNN + k] = bv[k]; mi[l*KNN + k] = bi[k]; }
  __syncthreads();
  if (l < 16){
    const float* va = &mv[l * KNN];        const int* ia = &mi[l * KNN];
    const float* vb = &mv[(l+16) * KNN];   const int* ib = &mi[(l+16) * KNN];
    int* out = idx + ((size_t)b * N_ + mrow) * KNN;
    int i = 0, j = 0;
    for (int k = 0; k < KNN; ++k){          // i+j==k<=19 at read time: in-bounds
      if (va[i] >= vb[j]) out[k] = ia[i++];
      else                out[k] = ib[j++];
    }
  }
}

// ---------------------------------------------------------------------------
// Kernel prep: f16 weight repacks (w2cat for P/Q factorization, w5h = W5).
// ---------------------------------------------------------------------------
__global__ __launch_bounds__(256) void k_prep(const float* __restrict__ W2, const float* __restrict__ W5,
                                              _Float16* __restrict__ w2cat, _Float16* __restrict__ w5h){
  for (int i = blockIdx.x * 256 + threadIdx.x; i < 1024 * 128; i += gridDim.x * 256){
    w5h[i] = (_Float16)W5[i];
    if (i < 128 * 64){
      int j = i >> 6, k = i & 63;
      float v = (j < 64) ? W2[j*128 + k]
                         : (W2[(j-64)*128 + 64 + k] - W2[(j-64)*128 + k]);
      w2cat[i] = (_Float16)v;
    }
  }
}

// ---------------------------------------------------------------------------
// Kernel 4a: PQ = X1 (Nx64 f16) @ w2cat^T (64x128) via WMMA.
// ---------------------------------------------------------------------------
__global__ __launch_bounds__(32) void k_pq(const _Float16* __restrict__ x12h,
                                           const _Float16* __restrict__ w2cat,
                                           float* __restrict__ PQ){
  int l  = threadIdx.x;
  int nt = blockIdx.x & 7;
  int mt = (blockIdx.x >> 3) & 127;
  int b  = blockIdx.x >> 10;
  const _Float16* xb = x12h + (size_t)b * N_ * 128;
  int mrow = (mt << 4) + (l & 15);
  const _Float16* arow = xb + (size_t)mrow * 128;
  int asel = (l >> 4) << 4;
  v16h a0 = cat2(*(const v8h*)(arow +  0 + asel), *(const v8h*)(arow +  8 + asel));
  v16h a1 = cat2(*(const v8h*)(arow + 32 + asel), *(const v8h*)(arow + 40 + asel));
  int col = (nt << 4) + (l & 15);
  const _Float16* brow = w2cat + (size_t)col * 64;
  int bsel = (l >> 4) << 4;
  v16h bt0 = *(const v16h*)(brow +  0 + bsel);
  v16h bt1 = *(const v16h*)(brow + 32 + bsel);
  v8f c = {};
  c = wmma_f16(a0, bt0, c);
  c = wmma_f16(a1, bt1, c);
  float* outb = PQ + (size_t)b * N_ * 128;
  int r0 = (l >> 4) << 3;
#pragma unroll
  for (int r = 0; r < 8; ++r)
    outb[(size_t)((mt << 4) + r0 + r) * 128 + col] = c[r];
}

// ---------------------------------------------------------------------------
// Kernel 4b: x2[n,o] = lrelu(Q[n,o] + max_k P[idx[n,k],o]).
// ---------------------------------------------------------------------------
__global__ __launch_bounds__(256) void k_edge2(const float* __restrict__ PQ, const int* __restrict__ idx,
                                               _Float16* __restrict__ x12h){
  int o  = threadIdx.x & 63;
  int pt = threadIdx.x >> 6;
  int b  = blockIdx.x >> 9;
  int n  = (blockIdx.x & 511) * 4 + pt;
  const float* pqb = PQ + (size_t)b * N_ * 128;
  const int* id = idx + ((size_t)b * N_ + n) * KNN;
  float best = -3.4e38f;
  for (int k = 0; k < KNN; ++k){
    int m = id[k];
    best = fmaxf(best, pqb[(size_t)m * 128 + o]);
  }
  float val = lrelu(best + pqb[(size_t)n * 128 + 64 + o]);
  x12h[((size_t)b * N_ + n) * 128 + 64 + o] = (_Float16)val;
}

// ---------------------------------------------------------------------------
// Kernel 5: h = lrelu(X12 @ W5^T) fused with per-batch column max+sum
// (h: 256MB never materialized). One wave per (batch, 32-column stripe):
// each wave holds TWO column-tiles of B resident (8 v16h) so every A
// fragment feeds 8 WMMAs -> halves L2 traffic vs 1 tile/wave. Next A row
// prefetched (global_prefetch_b8).
// ---------------------------------------------------------------------------
__global__ __launch_bounds__(32) void k_h(const _Float16* __restrict__ x12h,
                                          const _Float16* __restrict__ w5h,
                                          float* __restrict__ gmax, float* __restrict__ gsum){
  int l  = threadIdx.x;
  int cp = blockIdx.x & 31;             // column-pair stripe (32 cols)
  int b  = blockIdx.x >> 5;
  const _Float16* xb = x12h + (size_t)b * N_ * 128;
  int j0 = (cp << 5) + (l & 15);
  int j1 = j0 + 16;
  int bsel = (l >> 4) << 4;
  const _Float16* br0 = w5h + (size_t)j0 * 128;
  const _Float16* br1 = w5h + (size_t)j1 * 128;
  v16h B00 = *(const v16h*)(br0 +  0 + bsel);
  v16h B01 = *(const v16h*)(br0 + 32 + bsel);
  v16h B02 = *(const v16h*)(br0 + 64 + bsel);
  v16h B03 = *(const v16h*)(br0 + 96 + bsel);
  v16h B10 = *(const v16h*)(br1 +  0 + bsel);
  v16h B11 = *(const v16h*)(br1 + 32 + bsel);
  v16h B12 = *(const v16h*)(br1 + 64 + bsel);
  v16h B13 = *(const v16h*)(br1 + 96 + bsel);
  float cmax0[8], csum0[8], cmax1[8], csum1[8];
#pragma unroll
  for (int r = 0; r < 8; ++r){
    cmax0[r] = -3.4e38f; csum0[r] = 0.f;
    cmax1[r] = -3.4e38f; csum1[r] = 0.f;
  }
  int asel = (l >> 4) << 4;
  for (int mt = 0; mt < 128; ++mt){
    const _Float16* arow = xb + (size_t)((mt << 4) + (l & 15)) * 128;
    // warm next row tile (each lane-half covers one 128B line of its row)
    const _Float16* arow_n = xb + (size_t)((((mt + 1) & 127) << 4) + (l & 15)) * 128;
    __builtin_prefetch(arow_n + ((l >> 4) << 6), 0, 1);
    v16h a0 = cat2(*(const v8h*)(arow +  0 + asel), *(const v8h*)(arow +  8 + asel));
    v16h a1 = cat2(*(const v8h*)(arow + 32 + asel), *(const v8h*)(arow + 40 + asel));
    v16h a2 = cat2(*(const v8h*)(arow + 64 + asel), *(const v8h*)(arow + 72 + asel));
    v16h a3 = cat2(*(const v8h*)(arow + 96 + asel), *(const v8h*)(arow + 104 + asel));
    v8f c0 = {}, c1 = {};
    c0 = wmma_f16(a0, B00, c0);
    c0 = wmma_f16(a1, B01, c0);
    c0 = wmma_f16(a2, B02, c0);
    c0 = wmma_f16(a3, B03, c0);
    c1 = wmma_f16(a0, B10, c1);
    c1 = wmma_f16(a1, B11, c1);
    c1 = wmma_f16(a2, B12, c1);
    c1 = wmma_f16(a3, B13, c1);
#pragma unroll
    for (int r = 0; r < 8; ++r){
      float v0 = lrelu(c0[r]);
      cmax0[r] = fmaxf(cmax0[r], v0); csum0[r] += v0;
      float v1 = lrelu(c1[r]);
      cmax1[r] = fmaxf(cmax1[r], v1); csum1[r] += v1;
    }
  }
  float vmax0 = cmax0[0], vsum0 = csum0[0], vmax1 = cmax1[0], vsum1 = csum1[0];
#pragma unroll
  for (int r = 1; r < 8; ++r){
    vmax0 = fmaxf(vmax0, cmax0[r]); vsum0 += csum0[r];
    vmax1 = fmaxf(vmax1, cmax1[r]); vsum1 += csum1[r];
  }
  vmax0 = fmaxf(vmax0, __shfl_xor(vmax0, 16));
  vsum0 += __shfl_xor(vsum0, 16);
  vmax1 = fmaxf(vmax1, __shfl_xor(vmax1, 16));
  vsum1 += __shfl_xor(vsum1, 16);
  if (l < 16){
    gmax[(size_t)b * 1024 + j0] = vmax0;
    gsum[(size_t)b * 1024 + j0] = vsum0;
    gmax[(size_t)b * 1024 + j1] = vmax1;
    gsum[(size_t)b * 1024 + j1] = vsum1;
  }
}

// ---------------------------------------------------------------------------
// Kernel 6: feature branch f = lrelu(lrelu(xf@W3^T)@W4^T); per-batch max+sum.
// ---------------------------------------------------------------------------
__global__ __launch_bounds__(1024) void k_feat(const float* __restrict__ x, const float* __restrict__ W3,
                                               const float* __restrict__ W4,
                                               float* __restrict__ fmaxb, float* __restrict__ fsumb){
  __shared__ float w3[32 * 3];
  __shared__ float w4[64 * 32];
  __shared__ float rmax[1024], rsum[1024];
  int b = blockIdx.x;
  for (int i = threadIdx.x; i < 96; i += 1024) w3[i] = W3[i];
  for (int i = threadIdx.x; i < 2048; i += 1024) w4[i] = W4[i];
  __syncthreads();
  int o = threadIdx.x & 63;
  int g = threadIdx.x >> 6;            // 16 point groups
  const float* xb = x + (size_t)b * N_ * 6;
  float vmax = -3.4e38f, vsum = 0.f;
  for (int n = g; n < N_; n += 16){
    float f0 = xb[n*6+3], f1 = xb[n*6+4], f2 = xb[n*6+5];
    float acc = 0.f;
#pragma unroll
    for (int c = 0; c < 32; ++c){
      float h1 = lrelu(w3[c*3]*f0 + w3[c*3+1]*f1 + w3[c*3+2]*f2);
      acc += w4[o*32 + c] * h1;
    }
    float v = lrelu(acc);
    vmax = fmaxf(vmax, v); vsum += v;
  }
  rmax[threadIdx.x] = vmax; rsum[threadIdx.x] = vsum;
  __syncthreads();
  if (g == 0){
    for (int gg = 1; gg < 16; ++gg){
      vmax = fmaxf(vmax, rmax[gg*64 + o]);
      vsum += rsum[gg*64 + o];
    }
    fmaxb[(size_t)b * 64 + o] = vmax;
    fsumb[(size_t)b * 64 + o] = vsum;
  }
}

// ---------------------------------------------------------------------------
// Kernel 7: head MLP 2176 -> 128 -> 64 -> 40. One block per batch.
// ---------------------------------------------------------------------------
__global__ __launch_bounds__(128) void k_head(const float* __restrict__ gmax, const float* __restrict__ gsum,
                                              const float* __restrict__ fmaxb, const float* __restrict__ fsumb,
                                              const float* __restrict__ L1, const float* __restrict__ L2w,
                                              const float* __restrict__ L2b, const float* __restrict__ L3w,
                                              const float* __restrict__ L3b, float* __restrict__ out){
  __shared__ float z[2176];
  __shared__ float z1[128];
  __shared__ float z2[64];
  int b = blockIdx.x, t = threadIdx.x;
  const float invN = 1.f / (float)N_;
  for (int i = t; i < 2176; i += 128){
    float v;
    if      (i < 1024) v = gmax[(size_t)b*1024 + i];
    else if (i < 2048) v = gsum[(size_t)b*1024 + (i-1024)] * invN;
    else if (i < 2112) v = fmaxb[(size_t)b*64 + (i-2048)];
    else               v = fsumb[(size_t)b*64 + (i-2112)] * invN;
    z[i] = v;
  }
  __syncthreads();
  {
    const float* row = L1 + (size_t)t * 2176;
    float acc = 0.f;
    for (int i = 0; i < 2176; ++i) acc += row[i] * z[i];
    z1[t] = lrelu(acc);
  }
  __syncthreads();
  if (t < 64){
    const float* row = L2w + (size_t)t * 128;
    float acc = L2b[t];
    for (int i = 0; i < 128; ++i) acc += row[i] * z1[i];
    z2[t] = lrelu(acc);
  }
  __syncthreads();
  if (t < 40){
    const float* row = L3w + (size_t)t * 64;
    float acc = L3b[t];
    for (int i = 0; i < 64; ++i) acc += row[i] * z2[i];
    out[(size_t)b * 40 + t] = acc;
  }
}

// ---------------------------------------------------------------------------
extern "C" void kernel_launch(void* const* d_in, const int* in_sizes, int n_in,
                              void* d_out, int out_size, void* d_ws, size_t ws_size,
                              hipStream_t stream){
  const float* x   = (const float*)d_in[0];
  const float* W1  = (const float*)d_in[1];
  const float* W2  = (const float*)d_in[2];
  const float* W5  = (const float*)d_in[3];
  const float* W3  = (const float*)d_in[4];
  const float* W4  = (const float*)d_in[5];
  const float* L1  = (const float*)d_in[6];
  const float* L2w = (const float*)d_in[7];
  const float* L2b = (const float*)d_in[8];
  const float* L3w = (const float*)d_in[9];
  const float* L3b = (const float*)d_in[10];
  float* out = (float*)d_out;
  (void)in_sizes; (void)n_in; (void)out_size; (void)ws_size;

  char* p = (char*)d_ws;
  auto alloc = [&](size_t bytes) -> char* {
    char* r = p; p += (bytes + 255) & ~(size_t)255; return r;
  };
  int*      idx   = (int*)     alloc((size_t)B_ * N_ * KNN * 4);   // reused for both graphs
  _Float16* x12h  = (_Float16*)alloc((size_t)B_ * N_ * 128 * 2);   // x1 | x2 (f16)
  float*    sq1   = (float*)   alloc((size_t)B_ * N_ * 4);
  float*    PQ    = (float*)   alloc((size_t)B_ * N_ * 128 * 4);
  _Float16* w2cat = (_Float16*)alloc((size_t)128 * 64 * 2);
  _Float16* w5h   = (_Float16*)alloc((size_t)1024 * 128 * 2);
  float*    gmax  = (float*)   alloc((size_t)B_ * 1024 * 4);
  float*    gsum  = (float*)   alloc((size_t)B_ * 1024 * 4);
  float*    fmaxb = (float*)   alloc((size_t)B_ * 64 * 4);
  float*    fsumb = (float*)   alloc((size_t)B_ * 64 * 4);

  k_prep <<<dim3(512),        dim3(256),  0, stream>>>(W2, W5, w2cat, w5h);
  k_knn3 <<<dim3(B_ * 8),     dim3(256),  0, stream>>>(x, idx);
  k_edge1<<<dim3(B_ * 512),   dim3(256),  0, stream>>>(x, W1, idx, x12h, sq1);
  k_knn64<<<dim3(B_ * 128),   dim3(32),   0, stream>>>(x12h, sq1, idx);
  k_pq   <<<dim3(B_ * 128*8), dim3(32),   0, stream>>>(x12h, w2cat, PQ);
  k_edge2<<<dim3(B_ * 512),   dim3(256),  0, stream>>>(PQ, idx, x12h);
  k_h    <<<dim3(B_ * 32),    dim3(32),   0, stream>>>(x12h, w5h, gmax, gsum);
  k_feat <<<dim3(B_),         dim3(1024), 0, stream>>>(x, W3, W4, fmaxb, fsumb);
  k_head <<<dim3(B_),         dim3(128),  0, stream>>>(gmax, gsum, fmaxb, fsumb,
                                                       L1, L2w, L2b, L3w, L3b, out);
}
